// DenseGNNEncoder_36919538876774
// MI455X (gfx1250) — compile-verified, matrix operand order checked by reading, not run on hardware
//
#include <hip/hip_runtime.h>

#define N_NODES 100000
#define N_EDGES 800000
#define N_GRAPHS 128
#define IN_DIM 128
#define OUT_DIM 128
#define HID 32
#define EPS_BN 1e-5f

typedef __bf16 bf16_t;
typedef bf16_t  v16bf __attribute__((ext_vector_type(16)));
typedef float   v8f   __attribute__((ext_vector_type(8)));
typedef unsigned v8u  __attribute__((ext_vector_type(8)));
typedef unsigned short ush;

// ---------------------------------------------------------------- helpers

__device__ __forceinline__ unsigned bfbits(float f) {
  // round-to-nearest-even fp32 -> bf16 bits
  unsigned u = __float_as_uint(f);
  return (u + 0x7FFFu + ((u >> 16) & 1u)) >> 16;
}
__device__ __forceinline__ unsigned pack2bf(float lo, float hi) {
  return (bfbits(hi) << 16) | bfbits(lo);
}
__device__ __forceinline__ float bf2f(ush h) {
  return __uint_as_float(((unsigned)h) << 16);
}

// packed bf16 atomic add (2 elements) -- CDNA5 GLOBAL_ATOMIC_PK_ADD_BF16
__device__ __forceinline__ void atomic_pk_add_bf16(ush* p, unsigned v) {
  asm volatile("global_atomic_pk_add_bf16 %0, %1, off"
               :: "v"((unsigned long long)(uintptr_t)p), "v"(v) : "memory");
}

// A fragment: 16(M) x 32(K) bf16 from bf16 memory: 2x global_load_b128 per lane.
__device__ __forceinline__ v16bf load_a_bf(const ush* __restrict__ base,
                                           int ld, int row0, int k0) {
  int lane = threadIdx.x & 31;
  int m    = lane & 15;
  int kb   = (lane & 16) ? 8 : 0;
  const ush* rp = base + (size_t)(row0 + m) * ld + k0 + kb;
  uint4 q0 = *(const uint4*)(rp);        // k = k0+kb    .. +7
  uint4 q1 = *(const uint4*)(rp + 16);   // k = k0+kb+16 .. +23
  v8u t = {q0.x, q0.y, q0.z, q0.w, q1.x, q1.y, q1.z, q1.w};
  return __builtin_bit_cast(v16bf, t);
}

// B fragment from LDS, fragment-ready layout: 2x ds_load_b128.
// Index: ((((mtx*ks + s)*2 + nt)*32) + lane)*8 + v
__device__ __forceinline__ v16bf load_b_lds(const unsigned* ldsW, int ks,
                                            int mtx, int s, int nt) {
  int lane = threadIdx.x & 31;
  const uint4* fp =
      (const uint4*)(ldsW + (((((mtx * ks + s) << 1) + nt) << 5) + lane) * 8);
  uint4 a = fp[0];
  uint4 b = fp[1];
  v8u t = {a.x, a.y, a.z, a.w, b.x, b.y, b.z, b.w};
  return __builtin_bit_cast(v16bf, t);
}

// ---------------------------------------------------------------- kernels

// one-time fp32 -> bf16 matrix convert (network input x)
__global__ void f32_to_bf16_kernel(const float* __restrict__ src,
                                   ush* __restrict__ dst, long long n4) {
  long long stride = (long long)gridDim.x * blockDim.x;
  for (long long i = (long long)blockIdx.x * blockDim.x + threadIdx.x;
       i < n4; i += stride) {
    float4 v = *(const float4*)(src + 4 * i);
    uint2 w = {pack2bf(v.x, v.y), pack2bf(v.z, v.w)};
    *(uint2*)(dst + 4 * i) = w;
  }
}

// agg[dst[e]][c] += in[src[e]][c] ; bf16, 2 pk-bf16 atomics per 4 elements
__global__ void scatter_add_kernel(const ush* __restrict__ in, int ldin, int din,
                                   int kpad,
                                   const int* __restrict__ esrc,
                                   const int* __restrict__ edst,
                                   ush* __restrict__ agg) {
  int din4 = din >> 2;
  long long total  = (long long)N_EDGES * din4;
  long long stride = (long long)gridDim.x * blockDim.x;
  for (long long i = (long long)blockIdx.x * blockDim.x + threadIdx.x;
       i < total; i += stride) {
    int e = (int)(i / din4);
    int c = (int)(i % din4) << 2;
    int s = esrc[e];
    int d = edst[e];
    uint2 v = *(const uint2*)(in + (size_t)s * ldin + c);
    ush* ap = agg + (size_t)d * kpad + c;
    atomic_pk_add_bf16(ap,     v.x);
    atomic_pk_add_bf16(ap + 2, v.y);
  }
}

// out = agg @ Wr + bias + x @ Wl ; per-column sum/sumsq atomically accumulated.
// 256 threads = 8 wave32s; each wave a 16x32 output tile (2 N-tiles share A);
// software-pipelined A prefetch; weights staged fragment-ready in LDS (24 KB).
__global__ void __launch_bounds__(256)
gemm_wmma_kernel(const ush* __restrict__ agg, int kpad,
                 const ush* __restrict__ xin, int ldx,
                 const float* __restrict__ Wr, const float* __restrict__ Wl,
                 const float* __restrict__ bias,
                 int din, int dout,
                 float* __restrict__ out, float* __restrict__ stats) {
  __shared__ unsigned ldsW[2 * 6 * 2 * 32 * 8];  // [mtx][kstep][nt][lane][8], 24 KB

  int n0 = blockIdx.y * 32;
  int ks = kpad >> 5;                            // k-steps (<= 6)
  // stage bf16 zero-padded weights in per-lane fragment order
  int total = (2 * ks) << 9;
  for (int i = threadIdx.x; i < total; i += 256) {
    int v    = i & 7;
    int lane = (i >> 3) & 31;
    int nt   = (i >> 8) & 1;
    int rest = i >> 9;
    int s    = rest % ks;
    int mtx  = rest / ks;
    int n    = n0 + (nt << 4) + (lane & 15);
    int kb   = (lane & 16) ? 8 : 0;
    int k    = (s << 5) + ((v < 4) ? (kb + 2 * v) : (16 + kb + 2 * (v - 4)));
    const float* W = mtx ? Wl : Wr;
    float f0 = 0.f, f1 = 0.f;
    if (n < dout) {
      if (k < din)     f0 = W[(size_t)k * dout + n];
      if (k + 1 < din) f1 = W[(size_t)(k + 1) * dout + n];
    }
    ldsW[i] = pack2bf(f0, f1);
  }
  __syncthreads();

  int wave  = threadIdx.x >> 5;
  int mtile = blockIdx.x * 8 + wave;
  if (mtile * 16 >= N_NODES) return;       // wave-uniform, after the only barrier
  int row0 = mtile * 16;

  v8f c0 = {0.f, 0.f, 0.f, 0.f, 0.f, 0.f, 0.f, 0.f};
  v8f c1 = c0;
  // pipelined: prefetch next step's A fragments before consuming current ones
  // (one dummy over-read on the last step lands in the adjacent ws region)
  v16bf aA = load_a_bf(agg, kpad, row0, 0);
  v16bf aX = load_a_bf(xin, ldx, row0, 0);
  for (int s = 0; s < ks; ++s) {
    int k1 = (s + 1) << 5;
    v16bf aAn = load_a_bf(agg, kpad, row0, k1);
    v16bf aXn = load_a_bf(xin, ldx, row0, k1);
    v16bf bR0 = load_b_lds(ldsW, ks, 0, s, 0);
    v16bf bR1 = load_b_lds(ldsW, ks, 0, s, 1);
    v16bf bL0 = load_b_lds(ldsW, ks, 1, s, 0);
    v16bf bL1 = load_b_lds(ldsW, ks, 1, s, 1);
    c0 = __builtin_amdgcn_wmma_f32_16x16x32_bf16(false, aA, false, bR0,
                                                 (short)0, c0, false, false);
    c1 = __builtin_amdgcn_wmma_f32_16x16x32_bf16(false, aA, false, bR1,
                                                 (short)0, c1, false, false);
    c0 = __builtin_amdgcn_wmma_f32_16x16x32_bf16(false, aX, false, bL0,
                                                 (short)0, c0, false, false);
    c1 = __builtin_amdgcn_wmma_f32_16x16x32_bf16(false, aX, false, bL1,
                                                 (short)0, c1, false, false);
    aA = aAn;
    aX = aXn;
  }

  int lane  = threadIdx.x & 31;
  int rbase = row0 + ((lane & 16) ? 8 : 0);  // C layout: VGPR r -> M=r / M=8+r
#pragma unroll
  for (int nt = 0; nt < 2; ++nt) {
    int n = n0 + (nt << 4) + (lane & 15);
    if (n >= dout) continue;
    const v8f& c = nt ? c1 : c0;
    float bv = bias[n];
    float sum = 0.f, sq = 0.f;
#pragma unroll
    for (int r = 0; r < 8; ++r) {
      float v = c[r] + bv;
      out[(size_t)(rbase + r) * dout + n] = v;
      sum += v;
      sq  += v * v;
    }
    atomicAdd(&stats[2 * n],     sum);
    atomicAdd(&stats[2 * n + 1], sq);
  }
}

// y = gamma*(x-mu)*rsqrt(var+eps)+beta (+ReLU); f32 in, packed bf16 out
__global__ void bn_apply_kernel(const float* __restrict__ src, int dout,
                                const float* __restrict__ stats,
                                const float* __restrict__ gamma,
                                const float* __restrict__ beta,
                                ush* __restrict__ dst, int lddst, int dcol,
                                int relu) {
  int dout4 = dout >> 2;
  long long total  = (long long)N_NODES * dout4;
  long long stride = (long long)gridDim.x * blockDim.x;
  const float invN = 1.0f / (float)N_NODES;
  for (long long i = (long long)blockIdx.x * blockDim.x + threadIdx.x;
       i < total; i += stride) {
    int node = (int)(i / dout4);
    int col  = (int)(i % dout4) << 2;
    float4 v = *(const float4*)(src + (size_t)node * dout + col);
    float o[4] = {v.x, v.y, v.z, v.w};
#pragma unroll
    for (int j = 0; j < 4; ++j) {
      int cc = col + j;
      float mu  = stats[2 * cc] * invN;
      float var = stats[2 * cc + 1] * invN - mu * mu;
      float y = gamma[cc] * (o[j] - mu) * rsqrtf(var + EPS_BN) + beta[cc];
      o[j] = relu ? fmaxf(y, 0.0f) : y;
    }
    uint2 w = {pack2bf(o[0], o[1]), pack2bf(o[2], o[3])};
    *(uint2*)(dst + (size_t)node * lddst + dcol + col) = w;
  }
}

// segment-sum node features (bf16 -> f32 sums) + node counts per graph
__global__ void pool_accum_kernel(const ush* __restrict__ h, int ldh, int width,
                                  const int* __restrict__ batch,
                                  float* __restrict__ sums, float* __restrict__ cnt) {
  long long total  = (long long)N_NODES * width;
  long long stride = (long long)gridDim.x * blockDim.x;
  for (long long i = (long long)blockIdx.x * blockDim.x + threadIdx.x;
       i < total; i += stride) {
    int node = (int)(i / width);
    int col  = (int)(i % width);
    int b = batch[node];
    atomicAdd(&sums[(size_t)b * width + col], bf2f(h[(size_t)node * ldh + col]));
    if (col == 0) atomicAdd(&cnt[b], 1.0f);
  }
}

// out[g][o] = b[o] + sum_k (sums[g][k]/max(cnt,1)) * W[k][o]
__global__ void proj_kernel(const float* __restrict__ sums,
                            const float* __restrict__ cnt,
                            const float* __restrict__ W,
                            const float* __restrict__ bias,
                            float* __restrict__ out, int kdim) {
  int g = blockIdx.x;
  int o = threadIdx.x;
  float inv = 1.0f / fmaxf(cnt[g], 1.0f);
  float acc = bias[o];
  for (int k = 0; k < kdim; ++k)
    acc += sums[(size_t)g * kdim + k] * inv * W[(size_t)k * OUT_DIM + o];
  out[(size_t)g * OUT_DIM + o] = acc;
}

// ---------------------------------------------------------------- host driver

static void run_conv(hipStream_t stream,
                     const ush* in, int ldin, int din,
                     const float* Wr, const float* Wl, const float* bias,
                     const float* gamma, const float* beta, int dout,
                     ush* dst, int lddst, int dcol, int relu,
                     const int* esrc, const int* edst,
                     ush* AGG, float* OB, float* ST) {
  int kpad = (din + 31) & ~31;
  hipMemsetAsync(AGG, 0, (size_t)N_NODES * kpad * sizeof(ush), stream);
  hipMemsetAsync(ST, 0, (size_t)2 * dout * sizeof(float), stream);
  scatter_add_kernel<<<2048, 256, 0, stream>>>(in, ldin, din, kpad, esrc, edst, AGG);
  dim3 grid((N_NODES / 16 + 7) / 8, (dout + 31) / 32);
  gemm_wmma_kernel<<<grid, 256, 0, stream>>>(AGG, kpad, in, ldin, Wr, Wl, bias,
                                             din, dout, OB, ST);
  bn_apply_kernel<<<4096, 256, 0, stream>>>(OB, dout, ST, gamma, beta,
                                            dst, lddst, dcol, relu);
}

extern "C" void kernel_launch(void* const* d_in, const int* in_sizes, int n_in,
                              void* d_out, int out_size, void* d_ws, size_t ws_size,
                              hipStream_t stream) {
  (void)in_sizes; (void)n_in; (void)out_size;

  const float* x    = (const float*)d_in[0];
  const int*   esrc = (const int*)d_in[1];
  const int*   edst = ((const int*)d_in[1]) + N_EDGES;
  const int*   batch = (const int*)d_in[2];

  // workspace layout (bytes; all offsets 16B-aligned)
  const int CATLD = 192;
  const size_t CAT_B = 0;                                        // bf16 100000x192
  const size_t AGG_B = CAT_B + (size_t)N_NODES * CATLD * 2;      // bf16 100000x192
  const size_t XB_B  = AGG_B + (size_t)N_NODES * CATLD * 2;      // bf16 100000x128
  const size_t TB_B  = XB_B + (size_t)N_NODES * 128 * 2;         // bf16 100000x128
  const size_t OB_B  = TB_B + (size_t)N_NODES * 128 * 2;         // f32  100000x128
  const size_t ST_B  = OB_B + (size_t)N_NODES * 128 * 4;         // f32  256
  const size_t PO_B  = ST_B + 256 * 4;                           // f32  128x92
  const size_t CN_B  = PO_B + (size_t)N_GRAPHS * 92 * 4;         // f32  128
  const size_t NEEDED = CN_B + N_GRAPHS * 4;
  if (ws_size < NEEDED) return;

  char* w = (char*)d_ws;
  ush*   CAT = (ush*)(w + CAT_B);
  ush*   AGG = (ush*)(w + AGG_B);
  ush*   XB  = (ush*)(w + XB_B);
  ush*   TB  = (ush*)(w + TB_B);
  float* OB  = (float*)(w + OB_B);
  float* ST  = (float*)(w + ST_B);
  float* PO  = (float*)(w + PO_B);
  float* CN  = (float*)(w + CN_B);

  const float** P = (const float**)d_in;   // param leaves start at index 3
  static const int GROWTH[4] = {2, 3, 4, 4};

  // zero CAT once so padding columns read as finite zeros; convert x -> bf16
  hipMemsetAsync(CAT, 0, (size_t)N_NODES * CATLD * sizeof(ush), stream);
  f32_to_bf16_kernel<<<2048, 256, 0, stream>>>(x, XB,
                                               (long long)N_NODES * IN_DIM / 4);

  // stem: XB (ld 128, din 128) -> CAT[:, 0:32)
  run_conv(stream, XB, IN_DIM, IN_DIM,
           P[3], P[4], P[5], P[6], P[7], HID,
           CAT, CATLD, 0, /*relu=*/0, esrc, edst, AGG, OB, ST);

  int D = HID;
  int p = 8;
  for (int b = 0; b < 4; ++b) {
    int g = GROWTH[b];
    for (int i = 0; i < 3; ++i) {
      int din1 = D + HID * i;
      run_conv(stream, CAT, CATLD, din1,
               P[p], P[p + 1], P[p + 2], P[p + 3], P[p + 4], HID * g,
               TB, 128, 0, /*relu=*/1, esrc, edst, AGG, OB, ST);
      run_conv(stream, TB, 128, HID * g,
               P[p + 5], P[p + 6], P[p + 7], P[p + 8], P[p + 9], HID,
               CAT, CATLD, din1, /*relu=*/1, esrc, edst, AGG, OB, ST);
      p += 10;
    }
    int dinT = D + HID * 3;
    run_conv(stream, CAT, CATLD, dinT,
             P[p], P[p + 1], P[p + 2], P[p + 3], P[p + 4], dinT / 2,
             CAT, CATLD, 0, /*relu=*/0, esrc, edst, AGG, OB, ST);
    D = dinT / 2;
    p += 5;
  }
  // D == 92 ; h lives in CAT[:, 0:92) as bf16

  hipMemsetAsync(PO, 0, ((size_t)N_GRAPHS * 92 + N_GRAPHS) * sizeof(float), stream);
  pool_accum_kernel<<<2048, 256, 0, stream>>>(CAT, CATLD, D, batch, PO, CN);
  proj_kernel<<<N_GRAPHS, OUT_DIM, 0, stream>>>(PO, CN, P[p], P[p + 1],
                                                (float*)d_out, D);
}
